// SDRLayer_76768245449191
// MI455X (gfx1250) — matrix operand order repaired and checked by reading, not compiled, not run on hardware
//
#include <hip/hip_runtime.h>
#include <hip/hip_bf16.h>

#define N_NODES 100000
#define E_EDGES 1600000
#define DCH 64
#define NLISTS 3
#define EN_TOT (E_EDGES + N_NODES)   // edges + self loops
#define NEG_SLOPE 0.2f

typedef __attribute__((ext_vector_type(16))) __bf16 v16bf;
typedef __attribute__((ext_vector_type(8)))  __bf16 v8bf;
typedef __attribute__((ext_vector_type(8)))  float  v8f;

// ---------------------------------------------------------------------------
// Stage 0a: init accumulators
// ---------------------------------------------------------------------------
__global__ void init_norm_kernel(float* __restrict__ mx, float* __restrict__ denom) {
    int t = blockIdx.x * blockDim.x + threadIdx.x;
    if (t < NLISTS * N_NODES) {
        mx[t] = -__builtin_huge_valf();
        denom[t] = 0.f;
    }
}

__global__ void init_feats_kernel(const float* __restrict__ bias, float* __restrict__ feats) {
    size_t t = (size_t)blockIdx.x * blockDim.x + threadIdx.x;
    const size_t total = (size_t)NLISTS * N_NODES * DCH;
    if (t < total) {
        int l = (int)(t / ((size_t)N_NODES * DCH));
        int d = (int)(t % DCH);
        feats[t] = bias[l * DCH + d];
    }
}

// ---------------------------------------------------------------------------
// Stage 0b: pre-pack a row-major fp32 weight [K x 64] into bf16 B-fragment
// order.  Fragment f = (kstep, t): per lane 16 contiguous bf16 values,
//   pack[((ks*4 + t)*32 + lane)*16 + j] = W[(ks*32 + 8*(lane>>4) + (j&7) + 16*(j>>3))*64 + 16*t + (lane&15)]
// so the hot GEMM loop loads one contiguous 32B chunk per lane.
// ---------------------------------------------------------------------------
__global__ void pack_b_kernel(const float* __restrict__ W, __bf16* __restrict__ out, int K) {
    int idx = blockIdx.x * blockDim.x + threadIdx.x;
    int total = K * 64;                 // (K/32)*4*32*16
    if (idx >= total) return;
    int j    = idx & 15;
    int lane = (idx >> 4) & 31;
    int t    = (idx >> 9) & 3;
    int ks   = idx >> 11;
    int half = lane >> 4, c = lane & 15;
    out[idx] = (__bf16)W[(ks * 32 + 8 * half + (j & 7) + 16 * (j >> 3)) * DCH + 16 * t + c];
}

// ---------------------------------------------------------------------------
// helpers: A-fragment load (16x32 bf16) from a row-major fp32 row pointer.
// Runs of 8 floats are 32B-aligned -> float4 vector loads.
// ---------------------------------------------------------------------------
__device__ __forceinline__ v16bf load_a_frag(const float* __restrict__ rowk, int half) {
    const float4 p0 = *(const float4*)(rowk + 8 * half);
    const float4 p1 = *(const float4*)(rowk + 8 * half + 4);
    const float4 q0 = *(const float4*)(rowk + 16 + 8 * half);
    const float4 q1 = *(const float4*)(rowk + 16 + 8 * half + 4);
    v16bf a;
    a[0]  = (__bf16)p0.x; a[1]  = (__bf16)p0.y; a[2]  = (__bf16)p0.z; a[3]  = (__bf16)p0.w;
    a[4]  = (__bf16)p1.x; a[5]  = (__bf16)p1.y; a[6]  = (__bf16)p1.z; a[7]  = (__bf16)p1.w;
    a[8]  = (__bf16)q0.x; a[9]  = (__bf16)q0.y; a[10] = (__bf16)q0.z; a[11] = (__bf16)q0.w;
    a[12] = (__bf16)q1.x; a[13] = (__bf16)q1.y; a[14] = (__bf16)q1.z; a[15] = (__bf16)q1.w;
    return a;
}

__device__ __forceinline__ v16bf load_b_frag(const __bf16* __restrict__ pack,
                                             int kstep, int t, int lane) {
    return *(const v16bf*)(pack + (((kstep * 4 + t) * 32 + lane) << 4));
}

// ---------------------------------------------------------------------------
// Stage 1: H[l] = x @ W[l]  (bf16 WMMA, fp32 accum), plus a_s = H@att_src,
// a_d = H@att_dst reduced straight out of the WMMA accumulators.
// One wave owns a 16x64 tile (4 x v_wmma_f32_16x16x32_bf16 per k-step).
// ---------------------------------------------------------------------------
__global__ void __launch_bounds__(128)
gat_transform_kernel(const float* __restrict__ x,
                     const __bf16* __restrict__ Wp,     // packed [3][2*4*32*16]
                     const float* __restrict__ att_s,   // [3,64]
                     const float* __restrict__ att_d,   // [3,64]
                     __bf16* __restrict__ Hbf,          // [3,N,64]
                     float* __restrict__ a_s,           // [3,N]
                     float* __restrict__ a_d)           // [3,N]
{
    const int wave = threadIdx.x >> 5;
    const int lane = threadIdx.x & 31;
    const int half = lane >> 4;      // wave32: two 16-lane halves
    const int c    = lane & 15;
    const int l    = blockIdx.y;
    const int m0   = (blockIdx.x * 4 + wave) * 16;
    if (m0 >= N_NODES) return;       // wave-uniform: EXEC stays all-ones

    const int row    = m0 + c;
    const int rclamp = row < N_NODES ? row : (N_NODES - 1);
    const __bf16* Wl = Wp + l * (2 * 4 * 32 * 16);

    v8f acc[4];
    #pragma unroll
    for (int t = 0; t < 4; t++)
        #pragma unroll
        for (int v = 0; v < 8; v++) acc[t][v] = 0.f;

    #pragma unroll
    for (int ks = 0; ks < 2; ks++) {
        const v16bf a = load_a_frag(x + (size_t)rclamp * DCH + ks * 32, half);
        #pragma unroll
        for (int t = 0; t < 4; t++) {
            const v16bf b = load_b_frag(Wl, ks, t, lane);
            acc[t] = __builtin_amdgcn_wmma_f32_16x16x32_bf16(
                false, a, false, b, (short)0, acc[t], false, false);
        }
    }

    // attention vector slices for this lane's columns
    float asv[4], adv[4];
    #pragma unroll
    for (int t = 0; t < 4; t++) {
        asv[t] = att_s[l * DCH + 16 * t + c];
        adv[t] = att_d[l * DCH + 16 * t + c];
    }

    // C/D layout: VGPR v holds M = v + 8*half, N = 16t + c
    #pragma unroll
    for (int v = 0; v < 8; v++) {
        const int r = m0 + v + 8 * half;
        float ps = 0.f, pd = 0.f;
        #pragma unroll
        for (int t = 0; t < 4; t++) { ps += acc[t][v] * asv[t]; pd += acc[t][v] * adv[t]; }
        // reduce over the 16 lanes of this half (xor masks < 16 stay in-half)
        #pragma unroll
        for (int m = 1; m < 16; m <<= 1) {
            ps += __shfl_xor(ps, m, 32);
            pd += __shfl_xor(pd, m, 32);
        }
        if (r < N_NODES) {
            if (c == 0) {
                a_s[l * N_NODES + r] = ps;
                a_d[l * N_NODES + r] = pd;
            }
            __bf16* hp = Hbf + ((size_t)l * N_NODES + r) * DCH;
            #pragma unroll
            for (int t = 0; t < 4; t++) hp[16 * t + c] = (__bf16)acc[t][v];
        }
    }
}

// ---------------------------------------------------------------------------
// Stage 2: edge softmax (3 passes). Self-loops are edge ids >= E.
// ---------------------------------------------------------------------------
__device__ __forceinline__ void edge_endpoints(const int* __restrict__ eg, int t,
                                               int& s, int& d) {
    if (t < E_EDGES) { s = eg[t]; d = eg[E_EDGES + t]; }
    else             { s = d = t - E_EDGES; }
}

__device__ __forceinline__ float leaky(float e) {
    return e > 0.f ? e : NEG_SLOPE * e;
}

__device__ __forceinline__ void atomic_max_float(float* addr, float v) {
    // signed-max for non-negative, unsigned-min for negative (init = -inf)
    if (v >= 0.f) atomicMax((int*)addr, __float_as_int(v));
    else          atomicMin((unsigned int*)addr, __float_as_uint(v));
}

// pass 1: e = leaky(a_s[src] + a_d[dst]); store e; atomic segment-max by dst
__global__ void edge_max_kernel(const int* __restrict__ e0, const int* __restrict__ e1,
                                const int* __restrict__ e2,
                                const float* __restrict__ a_s, const float* __restrict__ a_d,
                                float* __restrict__ mx, float* __restrict__ ex)
{
    const int l = blockIdx.y;
    const int* eg = (l == 0) ? e0 : ((l == 1) ? e1 : e2);
    int t = blockIdx.x * blockDim.x + threadIdx.x;
    if (t >= EN_TOT) return;
    int s, d; edge_endpoints(eg, t, s, d);
    float e = leaky(a_s[l * N_NODES + s] + a_d[l * N_NODES + d]);
    ex[(size_t)l * EN_TOT + t] = e;           // streamed store; reused in pass 2
    atomic_max_float(mx + l * N_NODES + d, e);
}

// pass 2: ex = exp(e - max[dst]); atomic segment-sum by dst
__global__ void edge_expsum_kernel(const int* __restrict__ e0, const int* __restrict__ e1,
                                   const int* __restrict__ e2,
                                   const float* __restrict__ mx,
                                   float* __restrict__ ex, float* __restrict__ denom)
{
    const int l = blockIdx.y;
    const int* eg = (l == 0) ? e0 : ((l == 1) ? e1 : e2);
    int t = blockIdx.x * blockDim.x + threadIdx.x;
    if (t >= EN_TOT) return;
    int d = (t < E_EDGES) ? eg[E_EDGES + t] : (t - E_EDGES);
    float e = ex[(size_t)l * EN_TOT + t];     // streamed load (stored in pass 1)
    float v = __expf(e - mx[l * N_NODES + d]);
    ex[(size_t)l * EN_TOT + t] = v;
    atomicAdd(denom + l * N_NODES + d, v);
}

// pass 3: feats[dst] += alpha * H[src]; 64 threads per edge (one per feature)
__global__ void __launch_bounds__(256)
edge_aggregate_kernel(const int* __restrict__ e0, const int* __restrict__ e1,
                      const int* __restrict__ e2,
                      const __bf16* __restrict__ Hbf,
                      const float* __restrict__ ex, const float* __restrict__ denom,
                      float* __restrict__ feats)
{
    const int l = blockIdx.y;
    const int* eg = (l == 0) ? e0 : ((l == 1) ? e1 : e2);
    const int edge = blockIdx.x * 4 + (threadIdx.x >> 6);
    const int d    = threadIdx.x & 63;
    if (edge >= EN_TOT) return;
    int s, dst; edge_endpoints(eg, edge, s, dst);
    const float alpha = ex[(size_t)l * EN_TOT + edge] / denom[l * N_NODES + dst];
    const float hv = (float)Hbf[((size_t)l * N_NODES + s) * DCH + d];
    atomicAdd(feats + ((size_t)l * N_NODES + dst) * DCH + d, alpha * hv);
}

// ---------------------------------------------------------------------------
// Stage 3: fused concat + MLP.  combined = [x | f0 | f1 | f2]  (N x 256)
//   h1 = tanh(combined @ W1 + b1)   (WMMA, K=256)
//   out = h1 @ W2 + b2              (WMMA via LDS re-layout, K=64)
// ---------------------------------------------------------------------------
__global__ void __launch_bounds__(256)
mlp_kernel(const float* __restrict__ x, const float* __restrict__ feats,
           const __bf16* __restrict__ W1p,   // packed, K=256 (8 ksteps)
           const float* __restrict__ b1,
           const __bf16* __restrict__ W2p,   // packed, K=64 (2 ksteps)
           const float* __restrict__ b2,
           float* __restrict__ out)
{
    __shared__ __bf16 lds[8][16][DCH];   // per-wave 16x64 bf16 tile (16 KB)
    const int wave = threadIdx.x >> 5;
    const int lane = threadIdx.x & 31;
    const int half = lane >> 4;
    const int c    = lane & 15;
    const int m0   = (blockIdx.x * 8 + wave) * 16;
    const int row    = m0 + c;
    const int rclamp = row < N_NODES ? row : (N_NODES - 1);

    v8f acc[4];
    #pragma unroll
    for (int t = 0; t < 4; t++)
        #pragma unroll
        for (int v = 0; v < 8; v++) acc[t][v] = 0.f;

    #pragma unroll
    for (int ks = 0; ks < 8; ks++) {
        const int K0   = ks * 32;
        const int seg  = K0 >> 6;          // 64-col segment of combined
        const int koff = K0 & 63;          // [K0, K0+32) within one segment
        const float* src = (seg == 0)
            ? (x + (size_t)rclamp * DCH)
            : (feats + ((size_t)(seg - 1) * N_NODES + rclamp) * DCH);
        const v16bf a = load_a_frag(src + koff, half);
        #pragma unroll
        for (int t = 0; t < 4; t++) {
            const v16bf b = load_b_frag(W1p, ks, t, lane);
            acc[t] = __builtin_amdgcn_wmma_f32_16x16x32_bf16(
                false, a, false, b, (short)0, acc[t], false, false);
        }
    }

    // bias + tanh, stage to LDS for C->A fragment re-layout
    #pragma unroll
    for (int t = 0; t < 4; t++)
        #pragma unroll
        for (int v = 0; v < 8; v++)
            lds[wave][v + 8 * half][16 * t + c] =
                (__bf16)tanhf(acc[t][v] + b1[16 * t + c]);
    __syncthreads();

    v8f acc2[4];
    #pragma unroll
    for (int t = 0; t < 4; t++)
        #pragma unroll
        for (int v = 0; v < 8; v++) acc2[t][v] = 0.f;

    #pragma unroll
    for (int ks = 0; ks < 2; ks++) {
        const __bf16* lrow = &lds[wave][c][ks * 32];
        const v8bf a_lo = *(const v8bf*)(lrow + 8 * half);        // 16B aligned
        const v8bf a_hi = *(const v8bf*)(lrow + 16 + 8 * half);
        v16bf a;
        #pragma unroll
        for (int j = 0; j < 8; j++) { a[j] = a_lo[j]; a[8 + j] = a_hi[j]; }
        #pragma unroll
        for (int t = 0; t < 4; t++) {
            const v16bf b = load_b_frag(W2p, ks, t, lane);
            acc2[t] = __builtin_amdgcn_wmma_f32_16x16x32_bf16(
                false, a, false, b, (short)0, acc2[t], false, false);
        }
    }

    #pragma unroll
    for (int t = 0; t < 4; t++)
        #pragma unroll
        for (int v = 0; v < 8; v++) {
            const int r = m0 + v + 8 * half;
            if (r < N_NODES)
                out[(size_t)r * DCH + 16 * t + c] = acc2[t][v] + b2[16 * t + c];
        }
}

// ---------------------------------------------------------------------------
// Host launcher
// ---------------------------------------------------------------------------
extern "C" void kernel_launch(void* const* d_in, const int* in_sizes, int n_in,
                              void* d_out, int out_size, void* d_ws, size_t ws_size,
                              hipStream_t stream) {
    (void)in_sizes; (void)n_in; (void)out_size; (void)ws_size;

    const float* x       = (const float*)d_in[0];
    const int*   e0      = (const int*)d_in[1];
    const int*   e1      = (const int*)d_in[2];
    const int*   e2      = (const int*)d_in[3];
    const float* agg_W   = (const float*)d_in[4];
    const float* att_src = (const float*)d_in[5];
    const float* att_dst = (const float*)d_in[6];
    const float* agg_b   = (const float*)d_in[7];
    const float* W1      = (const float*)d_in[8];
    const float* b1      = (const float*)d_in[9];
    const float* W2      = (const float*)d_in[10];
    const float* b2      = (const float*)d_in[11];
    float*       out     = (float*)d_out;

    // workspace carve-up (256B aligned)
    char* p = (char*)d_ws;
    auto carve = [&](size_t bytes) -> void* {
        void* r = (void*)p;
        p += (bytes + 255) & ~(size_t)255;
        return r;
    };
    __bf16* Hbf   = (__bf16*)carve((size_t)NLISTS * N_NODES * DCH * sizeof(__bf16)); // 38.4 MB
    float*  a_s   = (float*)carve((size_t)NLISTS * N_NODES * sizeof(float));
    float*  a_d   = (float*)carve((size_t)NLISTS * N_NODES * sizeof(float));
    float*  mx    = (float*)carve((size_t)NLISTS * N_NODES * sizeof(float));
    float*  denom = (float*)carve((size_t)NLISTS * N_NODES * sizeof(float));
    float*  ex    = (float*)carve((size_t)NLISTS * EN_TOT * sizeof(float));          // 20.4 MB
    float*  feats = (float*)carve((size_t)NLISTS * N_NODES * DCH * sizeof(float));   // 76.8 MB
    __bf16* Wp    = (__bf16*)carve((size_t)NLISTS * 64 * 64 * sizeof(__bf16));       // packed agg_W
    __bf16* W1p   = (__bf16*)carve((size_t)256 * 64 * sizeof(__bf16));               // packed W1
    __bf16* W2p   = (__bf16*)carve((size_t)64 * 64 * sizeof(__bf16));                // packed W2

    // init + weight packing
    {
        int n = NLISTS * N_NODES;
        init_norm_kernel<<<(n + 255) / 256, 256, 0, stream>>>(mx, denom);
        size_t nf = (size_t)NLISTS * N_NODES * DCH;
        init_feats_kernel<<<(unsigned)((nf + 255) / 256), 256, 0, stream>>>(agg_b, feats);
        for (int l = 0; l < NLISTS; l++)
            pack_b_kernel<<<(64 * 64 + 255) / 256, 256, 0, stream>>>(
                agg_W + l * DCH * DCH, Wp + l * DCH * DCH, DCH);
        pack_b_kernel<<<(256 * 64 + 255) / 256, 256, 0, stream>>>(W1, W1p, 4 * DCH);
        pack_b_kernel<<<(64 * 64 + 255) / 256, 256, 0, stream>>>(W2, W2p, DCH);
    }

    // stage 1: dense transforms + attention scalars (all 3 lists via grid.y)
    {
        dim3 grid((N_NODES + 63) / 64, NLISTS);
        gat_transform_kernel<<<grid, 128, 0, stream>>>(x, Wp, att_src, att_dst,
                                                       Hbf, a_s, a_d);
    }

    // stage 2: edge softmax
    {
        dim3 grid((EN_TOT + 255) / 256, NLISTS);
        edge_max_kernel<<<grid, 256, 0, stream>>>(e0, e1, e2, a_s, a_d, mx, ex);
        edge_expsum_kernel<<<grid, 256, 0, stream>>>(e0, e1, e2, mx, ex, denom);
        dim3 grid3((EN_TOT + 3) / 4, NLISTS);
        edge_aggregate_kernel<<<grid3, 256, 0, stream>>>(e0, e1, e2, Hbf, ex, denom, feats);
    }

    // stage 3: fused concat + MLP
    {
        dim3 grid((N_NODES + 127) / 128);
        mlp_kernel<<<grid, 256, 0, stream>>>(x, feats, W1p, b1, W2p, b2, out);
    }
}
